// AttentionHeads_10264971837877
// MI455X (gfx1250) — matrix operand order repaired and checked by reading, not compiled, not run on hardware
//
#include <hip/hip_runtime.h>

typedef __attribute__((ext_vector_type(16))) __bf16 v16bf;
typedef __attribute__((ext_vector_type(8)))  __bf16 v8bf;
typedef __attribute__((ext_vector_type(8)))  float  v8f;
typedef __attribute__((ext_vector_type(4)))  unsigned int u32x4;
typedef __attribute__((ext_vector_type(8)))  int i32x8;
typedef __attribute__((ext_vector_type(4)))  int i32x4;

#define B_   2
#define T_   2048
#define EMB_ 2048
#define HQ_  16
#define HD_  128

static __device__ __forceinline__ v8f wmma_bf16(v16bf a, v16bf b, v8f c) {
  return __builtin_amdgcn_wmma_f32_16x16x32_bf16(false, a, false, b, (short)0, c,
                                                 false, false);
}

// A fragment: 16(M) x 32(K); lanes 0-15 row=lane K{0..7,16..23}, lanes 16-31 K{8..15,24..31}
static __device__ __forceinline__ v16bf load_a_frag(const __bf16* p, int ld) {
  const int l = threadIdx.x & 31;
  const __bf16* base = p + (size_t)(l & 15) * ld + ((l < 16) ? 0 : 8);
  v8bf lo = *(const v8bf*)(base);
  v8bf hi = *(const v8bf*)(base + 16);
  v16bf a;
#pragma unroll
  for (int i = 0; i < 8; ++i) { a[i] = lo[i]; a[8 + i] = hi[i]; }
  return a;
}

// B fragment: 32(K) x 16(N); each column's K-run contiguous at p[n*ld + k]
static __device__ __forceinline__ v16bf load_b_frag(const __bf16* p, int ld) {
  const int l = threadIdx.x & 31;
  const __bf16* base = p + (size_t)(l & 15) * ld + ((l < 16) ? 0 : 8);
  v8bf lo = *(const v8bf*)(base);
  v8bf hi = *(const v8bf*)(base + 16);
  v16bf b;
#pragma unroll
  for (int i = 0; i < 8; ++i) { b[i] = lo[i]; b[8 + i] = hi[i]; }
  return b;
}

// ---- Tensor Data Mover: 2-D bf16 tile (tile_d0 = contiguous axis) -> LDS
static __device__ __forceinline__ unsigned lds_off(const void* p) {
  return (unsigned)(unsigned long long)p;   // low 32 bits of generic addr = LDS offset
}

static __device__ __forceinline__ void tdm_load_2d(
    unsigned lds, const void* gaddr,
    unsigned tile_d0, unsigned tile_d1,
    unsigned tensor_d0, unsigned tensor_d1, unsigned stride_d0) {
  unsigned long long ga = (unsigned long long)gaddr;
  u32x4 g0;
  g0[0] = 1u;                                        // count=1, user descriptor
  g0[1] = lds;                                       // lds_addr (bytes)
  g0[2] = (unsigned)(ga & 0xFFFFFFFFu);              // global_addr[31:0]
  g0[3] = (unsigned)((ga >> 32) & 0x1FFFFFFu) | (2u << 30);  // addr[56:32] | type=2
  i32x8 g1;
  g1[0] = (int)(1u << 16);                           // data_size=1 (2 bytes)
  g1[1] = (int)((tensor_d0 & 0xFFFFu) << 16);        // tensor_dim0[15:0] @ 63:48
  g1[2] = (int)((tensor_d0 >> 16) | ((tensor_d1 & 0xFFFFu) << 16));
  g1[3] = (int)((tensor_d1 >> 16) | (tile_d0 << 16)); // tile_dim0 @ 127:112
  g1[4] = (int)(tile_d1 & 0xFFFFu);                   // tile_dim1; tile_dim2=0
  g1[5] = (int)stride_d0;                             // tensor_dim0_stride[31:0]
  g1[6] = 0;
  g1[7] = 0;
  i32x4 z4 = {0, 0, 0, 0};
  i32x8 z8 = {0, 0, 0, 0, 0, 0, 0, 0};
  __builtin_amdgcn_tensor_load_to_lds(g0, g1, z4, z4, z8, 0);
}

template <int N>
static __device__ __forceinline__ void tdm_wait() {
  __builtin_amdgcn_s_wait_tensorcnt(N);
}

// ---------------------------------------------------------------- convert
__global__ void cvt_bf16_kernel(const float* __restrict__ src,
                                __bf16* __restrict__ dst, int n) {
  int i = blockIdx.x * blockDim.x + threadIdx.x;
  if (i < n) dst[i] = (__bf16)src[i];
}

// ---------------------------------------------------------------- QKV + RoPE
// D[feature, token]. Block: 8 waves x 16 features, shared 64-token tile.
// x tile TDM-staged to LDS (double buffered); weight A-frags pipelined in regs.
__global__ void qkv_rope_kernel(const __bf16* __restrict__ xb,
                                const __bf16* __restrict__ wqb,
                                const __bf16* __restrict__ wkb,
                                const __bf16* __restrict__ wvb,
                                const float* __restrict__ fcos,
                                const float* __restrict__ fsin,
                                __bf16* __restrict__ qb,
                                __bf16* __restrict__ kb,
                                __bf16* __restrict__ vb) {
  __shared__ __align__(16) __bf16 xt[2][64][32];      // [buf][token][k]
  const int wave = threadIdx.x >> 5;
  const int lane = threadIdx.x & 31;
  const int tok0 = blockIdx.x * 64;
  const int f0   = blockIdx.y * 128 + wave * 16;
  const int proj = f0 / EMB_;
  const __bf16* w = (proj == 0) ? wqb : ((proj == 1) ? wkb : wvb);
  const int frow = f0 % EMB_;
  const int h    = frow / HD_;
  const bool issuer = (threadIdx.x < 32);

  if (issuer)
    tdm_load_2d(lds_off(&xt[0][0][0]), xb + (size_t)tok0 * EMB_,
                32, 64, EMB_, B_ * T_, EMB_);

  v8f zero = {};
  v8f acc[4];
#pragma unroll
  for (int j = 0; j < 4; ++j) acc[j] = zero;

  v16bf a_cur = load_a_frag(w + (size_t)frow * EMB_, EMB_);

  for (int i = 0; i < EMB_ / 32; ++i) {
    const int kk = i * 32;
    if (issuer) {
      if (i + 1 < EMB_ / 32) {
        tdm_load_2d(lds_off(&xt[(i + 1) & 1][0][0]),
                    xb + (size_t)tok0 * EMB_ + kk + 32,
                    32, 64, EMB_, B_ * T_, EMB_);
        tdm_wait<1>();
      } else {
        tdm_wait<0>();
      }
    }
    __syncthreads();
    v16bf a_next = a_cur;
    if (i + 1 < EMB_ / 32)
      a_next = load_a_frag(w + (size_t)frow * EMB_ + kk + 32, EMB_);
    const __bf16* Xt = &xt[i & 1][0][0];
#pragma unroll
    for (int j = 0; j < 4; ++j)
      acc[j] = wmma_bf16(a_cur, load_b_frag(Xt + j * 16 * 32, 32), acc[j]);
    a_cur = a_next;
    __syncthreads();
  }

  const int dbase = (f0 % HD_) + ((lane < 16) ? 0 : 8);
#pragma unroll
  for (int j = 0; j < 4; ++j) {
    const int tokrow = tok0 + j * 16 + (lane & 15);
    const int b = tokrow >> 11;
    const int t = tokrow & (T_ - 1);
    float v[8];
#pragma unroll
    for (int r = 0; r < 8; ++r) v[r] = acc[j][r];
    if (proj < 2) {
      const float* cs = fcos + (size_t)t * (HD_ / 2) + (dbase >> 1);
      const float* sn = fsin + (size_t)t * (HD_ / 2) + (dbase >> 1);
#pragma unroll
      for (int pr = 0; pr < 4; ++pr) {
        float c = cs[pr], s = sn[pr];
        float xr = v[2 * pr], xi = v[2 * pr + 1];
        v[2 * pr]     = xr * c - xi * s;
        v[2 * pr + 1] = xr * s + xi * c;
      }
    }
    if (proj == 2) {                                   // V -> [b,h,d,t]
      __bf16* dst = vb + ((size_t)(b * HQ_ + h) * HD_) * T_ + t;
#pragma unroll
      for (int r = 0; r < 8; ++r) dst[(size_t)(dbase + r) * T_] = (__bf16)v[r];
    } else {                                           // Q,K -> [b,h,t,d]
      __bf16* dst = ((proj == 0) ? qb : kb) +
                    ((size_t)(b * HQ_ + h) * T_ + t) * HD_ + dbase;
#pragma unroll
      for (int r = 0; r < 8; ++r) dst[r] = (__bf16)v[r];
    }
  }
}

// ---------------------------------------------------------------- flash attention
// grid (16, 32), block 256. K/V tiles TDM-staged (shared by all 8 waves).
__global__ void attn_kernel(const __bf16* __restrict__ qb,
                            const __bf16* __restrict__ kb,
                            const __bf16* __restrict__ vb,
                            __bf16* __restrict__ attnb) {
  __shared__ __align__(16) __bf16 kt[2][32][128];      // [buf][s][d]
  __shared__ __align__(16) __bf16 vt[2][128][32];      // [buf][d][s]
  __shared__ __align__(16) __bf16 p_lds[8][16][32];    // per-wave P [t][s]
  const int wave = threadIdx.x >> 5;
  const int lane = threadIdx.x & 31;
  const int bh = blockIdx.y;
  const int b = bh >> 4, h = bh & 15;
  const int t0 = blockIdx.x * 128 + wave * 16;
  const bool issuer = (threadIdx.x < 32);

  const __bf16* Q = qb + ((size_t)bh * T_ + t0) * HD_;
  const __bf16* K = kb + (size_t)bh * T_ * HD_;
  const __bf16* V = vb + (size_t)bh * HD_ * T_;

  if (issuer) {
    tdm_load_2d(lds_off(&kt[0][0][0]), K, 128, 32, HD_, T_, HD_);
    tdm_load_2d(lds_off(&vt[0][0][0]), V, 32, 128, T_, HD_, T_);
  }

  v16bf qf[4];
#pragma unroll
  for (int kd = 0; kd < 4; ++kd) qf[kd] = load_a_frag(Q + kd * 32, HD_);

  v8f zero = {};
  v8f o[8];
#pragma unroll
  for (int dt = 0; dt < 8; ++dt) o[dt] = zero;
  float mrow[8], lrow[8];
#pragma unroll
  for (int r = 0; r < 8; ++r) { mrow[r] = -1e30f; lrow[r] = 0.0f; }

  const float sc = 0.08838834764831845f;               // 1/sqrt(128)
  const int n  = lane & 15;
  const int mb = (lane < 16) ? 0 : 8;

  for (int i = 0; i < T_ / 32; ++i) {
    const int s0 = i * 32;
    if (issuer) {
      if (i + 1 < T_ / 32) {
        tdm_load_2d(lds_off(&kt[(i + 1) & 1][0][0]),
                    K + (size_t)(s0 + 32) * HD_, 128, 32, HD_, T_, HD_);
        tdm_load_2d(lds_off(&vt[(i + 1) & 1][0][0]),
                    V + (s0 + 32), 32, 128, T_, HD_, T_);
        tdm_wait<2>();
      } else {
        tdm_wait<0>();
      }
    }
    __syncthreads();
    const __bf16* Kt = &kt[i & 1][0][0];
    const __bf16* Vt = &vt[i & 1][0][0];

    v8f s1 = zero, s2 = zero;
#pragma unroll
    for (int kd = 0; kd < 4; ++kd) {
      s1 = wmma_bf16(qf[kd], load_b_frag(Kt + kd * 32, HD_), s1);
      s2 = wmma_bf16(qf[kd], load_b_frag(Kt + 16 * HD_ + kd * 32, HD_), s2);
    }
    float corr[8];
#pragma unroll
    for (int r = 0; r < 8; ++r) {
      float a = s1[r] * sc, c = s2[r] * sc;
      float mx = fmaxf(a, c);
      mx = fmaxf(mx, __shfl_xor(mx, 1, 32));
      mx = fmaxf(mx, __shfl_xor(mx, 2, 32));
      mx = fmaxf(mx, __shfl_xor(mx, 4, 32));
      mx = fmaxf(mx, __shfl_xor(mx, 8, 32));
      float mnew = fmaxf(mrow[r], mx);
      corr[r] = __expf(mrow[r] - mnew);
      mrow[r] = mnew;
      float p1 = __expf(a - mnew), p2 = __expf(c - mnew);
      s1[r] = p1; s2[r] = p2;
      float rs = p1 + p2;
      rs += __shfl_xor(rs, 1, 32);
      rs += __shfl_xor(rs, 2, 32);
      rs += __shfl_xor(rs, 4, 32);
      rs += __shfl_xor(rs, 8, 32);
      lrow[r] = lrow[r] * corr[r] + rs;
    }
#pragma unroll
    for (int dt = 0; dt < 8; ++dt)
#pragma unroll
      for (int r = 0; r < 8; ++r) o[dt][r] *= corr[r];
#pragma unroll
    for (int r = 0; r < 8; ++r) {
      p_lds[wave][mb + r][n]      = (__bf16)s1[r];
      p_lds[wave][mb + r][16 + n] = (__bf16)s2[r];
    }
    asm volatile("s_wait_dscnt 0" ::: "memory");
    v16bf pf = load_a_frag(&p_lds[wave][0][0], 32);
#pragma unroll
    for (int dt = 0; dt < 8; ++dt)
      o[dt] = wmma_bf16(pf, load_b_frag(Vt + dt * 16 * 32, 32), o[dt]);
    __syncthreads();
  }

  float invl[8];
#pragma unroll
  for (int r = 0; r < 8; ++r) invl[r] = 1.0f / lrow[r];
#pragma unroll
  for (int dt = 0; dt < 8; ++dt)
#pragma unroll
    for (int r = 0; r < 8; ++r)
      attnb[((size_t)(b * T_ + t0 + mb + r)) * EMB_ + h * HD_ + dt * 16 + n] =
          (__bf16)(o[dt][r] * invl[r]);
}

// ---------------------------------------------------------------- out projection
// Block: 8 waves share one 16-token A tile (TDM-staged); weight B pipelined in regs.
__global__ void oproj_kernel(const __bf16* __restrict__ attnb,
                             const __bf16* __restrict__ wob,
                             float* __restrict__ out) {
  __shared__ __align__(16) __bf16 at[2][16][32];
  const int wave = threadIdx.x >> 5;
  const int lane = threadIdx.x & 31;
  const int m0 = blockIdx.y * 16;
  const int n0 = blockIdx.x * 512 + wave * 64;
  const bool issuer = (threadIdx.x < 32);

  if (issuer)
    tdm_load_2d(lds_off(&at[0][0][0]), attnb + (size_t)m0 * EMB_,
                32, 16, EMB_, B_ * T_, EMB_);

  v8f zero = {};
  v8f acc[4];
#pragma unroll
  for (int j = 0; j < 4; ++j) acc[j] = zero;

  v16bf b_cur[4];
#pragma unroll
  for (int j = 0; j < 4; ++j)
    b_cur[j] = load_b_frag(wob + (size_t)(n0 + j * 16) * EMB_, EMB_);

  for (int i = 0; i < EMB_ / 32; ++i) {
    const int kk = i * 32;
    if (issuer) {
      if (i + 1 < EMB_ / 32) {
        tdm_load_2d(lds_off(&at[(i + 1) & 1][0][0]),
                    attnb + (size_t)m0 * EMB_ + kk + 32,
                    32, 16, EMB_, B_ * T_, EMB_);
        tdm_wait<1>();
      } else {
        tdm_wait<0>();
      }
    }
    __syncthreads();
    v16bf b_next[4];
#pragma unroll
    for (int j = 0; j < 4; ++j) b_next[j] = b_cur[j];
    if (i + 1 < EMB_ / 32) {
#pragma unroll
      for (int j = 0; j < 4; ++j)
        b_next[j] = load_b_frag(wob + (size_t)(n0 + j * 16) * EMB_ + kk + 32, EMB_);
    }
    v16bf a = load_a_frag(&at[i & 1][0][0], 32);
#pragma unroll
    for (int j = 0; j < 4; ++j) acc[j] = wmma_bf16(a, b_cur[j], acc[j]);
#pragma unroll
    for (int j = 0; j < 4; ++j) b_cur[j] = b_next[j];
    __syncthreads();
  }
  const int n  = lane & 15;
  const int mb = (lane < 16) ? 0 : 8;
#pragma unroll
  for (int j = 0; j < 4; ++j)
#pragma unroll
    for (int r = 0; r < 8; ++r)
      out[(size_t)(m0 + mb + r) * EMB_ + (n0 + j * 16 + n)] = acc[j][r];
}

// ---------------------------------------------------------------- launch
extern "C" void kernel_launch(void* const* d_in, const int* in_sizes, int n_in,
                              void* d_out, int out_size, void* d_ws, size_t ws_size,
                              hipStream_t stream) {
  const float* x    = (const float*)d_in[0];
  const float* wq   = (const float*)d_in[1];
  const float* wk   = (const float*)d_in[2];
  const float* wv   = (const float*)d_in[3];
  const float* wo   = (const float*)d_in[4];
  const float* fcos = (const float*)d_in[7];
  const float* fsin = (const float*)d_in[8];
  float* out = (float*)d_out;

  const size_t NX = (size_t)B_ * T_ * EMB_;
  const size_t NW = (size_t)EMB_ * EMB_;

  __bf16* p     = (__bf16*)d_ws;
  __bf16* xb    = p; p += NX;
  __bf16* wqb   = p; p += NW;
  __bf16* wkb   = p; p += NW;
  __bf16* wvb   = p; p += NW;
  __bf16* wob   = p; p += NW;
  __bf16* qb    = p; p += NX;
  __bf16* kb    = p; p += NX;
  __bf16* vb    = p; p += NX;
  __bf16* attnb = p; p += NX;

  dim3 blk(256);
  cvt_bf16_kernel<<<dim3((unsigned)((NX + 255) / 256)), blk, 0, stream>>>(x,  xb,  (int)NX);
  cvt_bf16_kernel<<<dim3((unsigned)((NW + 255) / 256)), blk, 0, stream>>>(wq, wqb, (int)NW);
  cvt_bf16_kernel<<<dim3((unsigned)((NW + 255) / 256)), blk, 0, stream>>>(wk, wkb, (int)NW);
  cvt_bf16_kernel<<<dim3((unsigned)((NW + 255) / 256)), blk, 0, stream>>>(wv, wvb, (int)NW);
  cvt_bf16_kernel<<<dim3((unsigned)((NW + 255) / 256)), blk, 0, stream>>>(wo, wob, (int)NW);

  qkv_rope_kernel<<<dim3(64, 48), blk, 0, stream>>>(xb, wqb, wkb, wvb,
                                                    fcos, fsin, qb, kb, vb);
  attn_kernel<<<dim3(16, 32), blk, 0, stream>>>(qb, kb, vb, attnb);
  oproj_kernel<<<dim3(4, 256), blk, 0, stream>>>(attnb, wob, out);
}
